// DepModule_4_68075231642183
// MI455X (gfx1250) — matrix operand order repaired and checked by reading, not compiled, not run on hardware
//
#include <hip/hip_runtime.h>
#include <math.h>

// ---------------------------------------------------------------------------
// CDNA5 / gfx1250 implementation.
// Heavy convs -> fused implicit-GEMM using v_wmma_f32_16x16x32_f16 (wave32),
// tap-outer K loop (hoists reflect / deform descriptor work), 2 WMMAs per
// wave per K-step (16x32 tile per wave, 32x128 macro-tile per 8-wave block).
// Light ops (deform bilinear descriptors, SE, BN stats, upsample) -> VALU.
// ---------------------------------------------------------------------------

typedef _Float16 h16;
typedef _Float16 v16h __attribute__((ext_vector_type(16)));
typedef float    v8f  __attribute__((ext_vector_type(8)));
typedef _Float16 h4   __attribute__((ext_vector_type(4)));

union V16H { v16h v; h16  e[16]; };
union V8F  { v8f  v; float e[8];  };

#define ACT_NONE 0
#define ACT_RELU 1
#define ACT_ELU  2
#define ACT_SIG  3

#define MODE_CONV3 0   // reflect-pad 3x3 conv
#define MODE_CONV1 1   // 1x1 conv
#define MODE_DEF   2   // deformable 3x3 conv (bilinear descriptors)

__device__ __forceinline__ float act_apply(float v, int act) {
  if (act == ACT_RELU) return fmaxf(v, 0.0f);
  if (act == ACT_ELU)  return v > 0.0f ? v : (expf(v) - 1.0f);
  if (act == ACT_SIG)  return 1.0f / (1.0f + expf(-v));
  return v;
}

// ----- weight repack: fp32 OIHW -> f16 [M][K], K tap-major (k = tap*Cin+ci) --
__global__ void repack_w_kernel(const float* __restrict__ w, h16* __restrict__ o,
                                int M, int Cin, int khw) {
  int K = Cin * khw;
  int idx = blockIdx.x * 256 + threadIdx.x;
  if (idx >= M * K) return;
  int m = idx / K, r = idx - m * K;
  int tap = r / Cin, ci = r - tap * Cin;
  o[idx] = (h16)w[(m * Cin + ci) * khw + tap];
}

// ----- deformable bilinear descriptors: 4 clamped indices + masked weights ---
#define CORNER(res_i, res_w, DY, DX) {                                  \
    int iy = iy0 + (DY), ix = ix0 + (DX);                               \
    float wv = ((DY) ? fy : 1.0f - fy) * ((DX) ? fx : 1.0f - fx);       \
    bool ok = (iy >= 0) && (iy < H) && (ix >= 0) && (ix < W);           \
    int iyc = iy < 0 ? 0 : (iy > H - 1 ? H - 1 : iy);                   \
    int ixc = ix < 0 ? 0 : (ix > W - 1 ? W - 1 : ix);                   \
    (res_i) = iyc * W + ixc;                                            \
    (res_w) = ok ? wv : 0.0f; }

__global__ void deform_desc_kernel(const float* __restrict__ off,
                                   int4* __restrict__ didx, float4* __restrict__ dwgt,
                                   int B, int H, int W) {
  int HW = H * W;
  long i = (long)blockIdx.x * 256 + threadIdx.x;
  if (i >= (long)B * 9 * HW) return;
  int p   = (int)(i % HW);
  int t   = (int)(i / HW);
  int tap = t % 9, b = t / 9;
  int h = p / W, w = p - h * W;
  int ki = tap / 3, kj = tap - ki * 3;
  float dy = off[((long)(b * 18 + 2 * tap    )) * HW + p];
  float dx = off[((long)(b * 18 + 2 * tap + 1)) * HW + p];
  float y = (float)(h - 1 + ki) + dy;
  float x = (float)(w - 1 + kj) + dx;
  float y0 = floorf(y), x0 = floorf(x);
  float fy = y - y0,    fx = x - x0;
  int iy0 = (int)y0, ix0 = (int)x0;
  int4 id; float4 wg;
  CORNER(id.x, wg.x, 0, 0);
  CORNER(id.y, wg.y, 0, 1);
  CORNER(id.z, wg.z, 1, 0);
  CORNER(id.w, wg.w, 1, 1);
  didx[i] = id;
  dwgt[i] = wg;
}

// ----- fused implicit-GEMM conv, WMMA f32 = f16 x f16 + f32 ------------------
// C[M,N] = W[M,K] * im2col[K,N], N = B*H*W, K = khw*Cin (tap-major).
// Block: 256 threads = 8 wave32s as 2(M) x 4(N); macro tile 32x128;
// each wave owns a 16x32 output tile -> 2 WMMAs per K-step sharing one A frag.
// Loop nest: taps outer (hoists reflect/deform setup), Cin chunks of 32 inner.
template <int MODE, int LOGW>
__global__ void __launch_bounds__(256)
wmma_conv_kernel(const h16*  __restrict__ A,      // [M,K] tap-major f16
                 const float* __restrict__ X,     // NCHW input
                 const float* __restrict__ bias,  // [M]
                 float* __restrict__ out,         // NCHW slice of Ctot channels
                 const int4*   __restrict__ didx,
                 const float4* __restrict__ dwgt,
                 int M, int Cin, int khw, int act, int Ctot, int c_off) {
  constexpr int W  = 1 << LOGW;
  constexpr int H  = W;
  constexpr int HW = H * W;
  const int K = Cin * khw;

  __shared__ __align__(16) h16 As[32 * 32];    // 2 KB
  __shared__ __align__(16) h16 Bs[128 * 32];   // 8 KB

  const int tid  = threadIdx.x;
  const int lane = tid & 31;
  const int wave = tid >> 5;
  const int wm = wave >> 2, wn = wave & 3;     // 2x4 waves over 32x128 tile
  const int blockN = blockIdx.x * 128;
  const int blockM = blockIdx.y * 32;

  // B-tile loader coords: 2 threads per pixel row, 16 k-elements each
  const int nloc  = tid >> 1;                  // 0..127
  const int kloc0 = (tid & 1) * 16;            // 0,16
  const int n = blockN + nloc;
  const int b = n >> (2 * LOGW);
  const int p = n & (HW - 1);
  const int h = p >> LOGW;
  const int w = p & (W - 1);

  // A-tile loader coords: 8 threads per weight row, 4 halves each
  const int ar = tid >> 3;                     // 0..31
  const int ac = (tid & 7) * 4;                // 0..28
  const h16* Arow = A + (long)(blockM + ar) * K + ac;

  V8F acc0, acc1;
#pragma unroll
  for (int r = 0; r < 8; ++r) { acc0.e[r] = 0.0f; acc1.e[r] = 0.0f; }

  int ks = 0;
  for (int tap = 0; tap < khw; ++tap) {
    // ---- per-tap setup, hoisted out of the channel-chunk loop ----
    const float* xbase;
    float4 wg; int4 id;
    if constexpr (MODE == MODE_CONV3) {
      const int ki = tap / 3, kj = tap - ki * 3;
      int iy = h - 1 + ki; iy = iy < 0 ? -iy : (iy >= H ? 2 * H - 2 - iy : iy);
      int ix = w - 1 + kj; ix = ix < 0 ? -ix : (ix >= W ? 2 * W - 2 - ix : ix);
      xbase = X + ((long)b * Cin) * HW + iy * W + ix;
    } else if constexpr (MODE == MODE_CONV1) {
      xbase = X + ((long)b * Cin) * HW + p;
    } else {
      const long d = (long)(b * 9 + tap) * HW + p;
      wg = dwgt[d];
      id = didx[d];
      xbase = X + (long)b * Cin * HW;
    }

    for (int c0 = 0; c0 < Cin; c0 += 32, ks += 32) {
      // ---- stage A tile (pre-packed f16 weights) + prefetch next chunk ----
      *(h4*)&As[ar * 32 + ac] = *(const h4*)&Arow[ks];
      __builtin_prefetch(&Arow[ks + 32], 0, 1);   // global_prefetch_b8

      // ---- stage B tile (im2col on the fly), 16 channels per thread ----
      const float* xc = xbase + (long)(c0 + kloc0) * HW;
      if constexpr (MODE == MODE_DEF) {
#pragma unroll
        for (int i = 0; i < 16; ++i) {
          float v = wg.x * xc[id.x] + wg.y * xc[id.y] +
                    wg.z * xc[id.z] + wg.w * xc[id.w];
          Bs[nloc * 32 + kloc0 + i] = (h16)v;
          xc += HW;
        }
      } else {
#pragma unroll
        for (int i = 0; i < 16; ++i) {
          Bs[nloc * 32 + kloc0 + i] = (h16)(*xc);
          xc += HW;
        }
      }
      __syncthreads();

      // ---- fragments: contiguous k order, k = 16*(lane/16) + s, same in A and B ----
      V16H a, b0, b1;
      const int mrow = wm * 16 + (lane & 15);
      const int kh = (lane >> 4) * 16; // A: lane<16 -> K 0-15, else K 16-31
#pragma unroll
      for (int i = 0; i < 8; ++i) {
        a.e[i]     = As[mrow * 32 + kh + i];
        a.e[8 + i] = As[mrow * 32 + kh + 8 + i];
      }
      const int kb  = (lane >> 4) * 16; // B: lane<16 -> K 0-15, else K 16-31
      const int nr0 = wn * 32 + (lane & 15);
#pragma unroll
      for (int i = 0; i < 16; ++i) {
        b0.e[i] = Bs[nr0 * 32 + kb + i];
        b1.e[i] = Bs[(nr0 + 16) * 32 + kb + i];
      }
      acc0.v = __builtin_amdgcn_wmma_f32_16x16x32_f16(
          false, a.v, false, b0.v, (short)0, acc0.v, false, false);
      acc1.v = __builtin_amdgcn_wmma_f32_16x16x32_f16(
          false, a.v, false, b1.v, (short)0, acc1.v, false, false);
      __syncthreads();
    }
  }

  // ---- epilogue: D layout -> VGPR r holds M = r + 8*(lane>=16), N = lane&15
  const int mb = blockM + wm * 16 + ((lane >> 4) << 3);
#pragma unroll
  for (int t = 0; t < 2; ++t) {
    const V8F* accp = t ? &acc1 : &acc0;
    const int n_out = blockN + wn * 32 + t * 16 + (lane & 15);
    const int ob = n_out >> (2 * LOGW);
    const int op = n_out & (HW - 1);
#pragma unroll
    for (int r = 0; r < 8; ++r) {
      int m = mb + r;
      float v = act_apply(accp->e[r] + bias[m], act);
      out[((long)ob * Ctot + c_off + m) * HW + op] = v;
    }
  }
}

// ----- SE: global average pool over HW per (b,c) -----------------------------
__global__ void se_pool_kernel(const float* __restrict__ x, float* __restrict__ g,
                               int C, int HW, int Ctot, int c_off) {
  int bc = blockIdx.x;
  int b = bc / C, c = bc - b * C;
  const float* p = x + ((long)b * Ctot + c_off + c) * HW;
  float s = 0.0f;
  for (int i = threadIdx.x; i < HW; i += 256) s += p[i];
  __shared__ float red[256];
  red[threadIdx.x] = s;
  __syncthreads();
  for (int st = 128; st > 0; st >>= 1) {
    if (threadIdx.x < st) red[threadIdx.x] += red[threadIdx.x + st];
    __syncthreads();
  }
  if (threadIdx.x == 0) g[bc] = red[0] / (float)HW;
}

// ----- SE: tiny per-batch FC (C x C), relu or sigmoid ------------------------
__global__ void se_fc_kernel(const float* __restrict__ g, const float* __restrict__ w,
                             const float* __restrict__ bias, float* __restrict__ o,
                             int Bn, int C, int act) {
  int i = blockIdx.x * 256 + threadIdx.x;
  if (i >= Bn * C) return;
  int b = i / C, co = i - b * C;
  const float* gr = g + (long)b * C;
  const float* wr = w + (long)co * C;
  float s = bias[co];
  for (int ci = 0; ci < C; ++ci) s += wr[ci] * gr[ci];
  o[i] = act_apply(s, act);
}

// ----- SE: out = out + out * a[b,c] -----------------------------------------
__global__ void se_apply_kernel(float* __restrict__ x, const float* __restrict__ a,
                                int C, int HW, int Ctot, int c_off, long total) {
  long i = (long)blockIdx.x * 256 + threadIdx.x;
  if (i >= total) return;
  int p = (int)(i % HW);
  long t = i / HW;
  int c = (int)(t % C);
  int b = (int)(t / C);
  float* q = x + ((long)b * Ctot + c_off + c) * HW + p;
  float v = *q;
  *q = v + v * a[b * C + c];
}

// ----- BN training-mode batch stats (per channel over B,HW) ------------------
__global__ void bn_stats_kernel(const float* __restrict__ x, float* __restrict__ mean,
                                float* __restrict__ inv, int C, int HW, int B) {
  int c = blockIdx.x;
  float s = 0.0f, s2 = 0.0f;
  for (int b = 0; b < B; ++b) {
    const float* p = x + ((long)b * C + c) * HW;
    for (int i = threadIdx.x; i < HW; i += 256) { float v = p[i]; s += v; s2 += v * v; }
  }
  __shared__ float r1[256], r2[256];
  r1[threadIdx.x] = s; r2[threadIdx.x] = s2;
  __syncthreads();
  for (int st = 128; st > 0; st >>= 1) {
    if (threadIdx.x < st) { r1[threadIdx.x] += r1[threadIdx.x + st];
                            r2[threadIdx.x] += r2[threadIdx.x + st]; }
    __syncthreads();
  }
  if (threadIdx.x == 0) {
    float n = (float)B * (float)HW;
    float m = r1[0] / n;
    float var = r2[0] / n - m * m;
    mean[c] = m;
    inv[c]  = rsqrtf(fmaxf(var, 0.0f) + 1e-5f);
  }
}

__global__ void bn_apply_relu_kernel(float* __restrict__ x,
                                     const float* __restrict__ mean,
                                     const float* __restrict__ inv,
                                     const float* __restrict__ g,
                                     const float* __restrict__ bb,
                                     int C, int HW, long total) {
  long i = (long)blockIdx.x * 256 + threadIdx.x;
  if (i >= total) return;
  int c = (int)((i / HW) % C);
  float v = (x[i] - mean[c]) * inv[c] * g[c] + bb[c];
  x[i] = fmaxf(v, 0.0f);
}

// ----- 2x nearest upsample into channel slice --------------------------------
__global__ void upsample2x_kernel(const float* __restrict__ in, float* __restrict__ out,
                                  int B, int C, int Hs, int Ws, int Ctot, int c_off) {
  int Hd = Hs * 2, Wd = Ws * 2;
  long HWd = (long)Hd * Wd;
  long i = (long)blockIdx.x * 256 + threadIdx.x;
  if (i >= (long)B * C * HWd) return;
  int x = (int)(i % Wd);
  long t = i / Wd;
  int y = (int)(t % Hd); t /= Hd;
  int c = (int)(t % C);
  int b = (int)(t / C);
  out[((long)b * Ctot + c_off + c) * HWd + (long)y * Wd + x] =
      in[((long)b * C + c) * (Hs * Ws) + (y >> 1) * Ws + (x >> 1)];
}

// ----- channel-slice copy (for concat) ---------------------------------------
__global__ void copy_ch_kernel(const float* __restrict__ in, float* __restrict__ out,
                               int B, int C, int HW, int Ctot, int c_off) {
  long i = (long)blockIdx.x * 256 + threadIdx.x;
  if (i >= (long)B * C * HW) return;
  int p = (int)(i % HW);
  long t = i / HW;
  int c = (int)(t % C);
  int b = (int)(t / C);
  out[((long)b * Ctot + c_off + c) * HW + p] = in[((long)b * C + c) * HW + p];
}

// ---------------------------------------------------------------------------

extern "C" void kernel_launch(void* const* d_in, const int* in_sizes, int n_in,
                              void* d_out, int out_size, void* d_ws, size_t ws_size,
                              hipStream_t stream) {
  (void)in_sizes; (void)n_in; (void)out_size; (void)ws_size;
  // Inputs flattened in setup_inputs() insertion order (params dict recursive).
  const float* input_feature = (const float*)d_in[0];   // [4,256,64,64]
  const float* offset_small  = (const float*)d_in[1];   // [4,18,64,64]
  const float* encoder       = (const float*)d_in[2];   // [4,128,128,128]
  const float* offset_large  = (const float*)d_in[3];   // [4,18,128,128]
  const float* conv1_w = (const float*)d_in[4];  const float* conv1_b = (const float*)d_in[5];
  const float* conv2_w = (const float*)d_in[6];  const float* conv2_b = (const float*)d_in[7];
  const float* conv3_w = (const float*)d_in[8];  const float* conv3_b = (const float*)d_in[9];
  const float* aug1_w  = (const float*)d_in[10]; const float* aug1_b  = (const float*)d_in[11];
  const float* aug2_w  = (const float*)d_in[12]; const float* aug2_b  = (const float*)d_in[13];
  const float* aug3_w  = (const float*)d_in[14]; const float* aug3_b  = (const float*)d_in[15];
  // att1 (C=256)
  const float* a1_cw  = (const float*)d_in[16]; const float* a1_cb  = (const float*)d_in[17];
  const float* a1_f1w = (const float*)d_in[18]; const float* a1_f1b = (const float*)d_in[19];
  const float* a1_f2w = (const float*)d_in[20]; const float* a1_f2b = (const float*)d_in[21];
  // att2 (C=192)
  const float* a2_cw  = (const float*)d_in[22]; const float* a2_cb  = (const float*)d_in[23];
  const float* a2_f1w = (const float*)d_in[24]; const float* a2_f1b = (const float*)d_in[25];
  const float* a2_f2w = (const float*)d_in[26]; const float* a2_f2b = (const float*)d_in[27];
  // att3 (C=448) + BN
  const float* a3_cw  = (const float*)d_in[28]; const float* a3_cb  = (const float*)d_in[29];
  const float* a3_f1w = (const float*)d_in[30]; const float* a3_f1b = (const float*)d_in[31];
  const float* a3_f2w = (const float*)d_in[32]; const float* a3_f2b = (const float*)d_in[33];
  const float* a3_bng = (const float*)d_in[34]; const float* a3_bnb = (const float*)d_in[35];
  // att4 (C=352)
  const float* a4_cw  = (const float*)d_in[36]; const float* a4_cb  = (const float*)d_in[37];
  const float* a4_f1w = (const float*)d_in[38]; const float* a4_f1b = (const float*)d_in[39];
  const float* a4_f2w = (const float*)d_in[40]; const float* a4_f2b = (const float*)d_in[41];

  const int B = 4;
  const int HW1 = 64 * 64;     // small resolution (LOGW=6)
  const int HW2 = 128 * 128;   // large resolution (LOGW=7)

  // ---- workspace bump allocator ----
  char* base = (char*)d_ws;
  size_t off = 0;
  auto alloc = [&](size_t bytes) -> void* {
    void* p = base + off;
    off += (bytes + 255) & ~(size_t)255;
    return p;
  };

  h16* wq_conv1 = (h16*)alloc((size_t)128 * 2304 * 2);
  h16* wq_aug1  = (h16*)alloc((size_t)128 * 2304 * 2);
  h16* wq_att1  = (h16*)alloc((size_t)256 * 2304 * 2);
  h16* wq_aug2  = (h16*)alloc((size_t)64  * 1152 * 2);
  h16* wq_att2  = (h16*)alloc((size_t)192 * 1728 * 2);
  h16* wq_att3  = (h16*)alloc((size_t)448 * 4032 * 2);
  h16* wq_aug3  = (h16*)alloc((size_t)224 * 4032 * 2);
  h16* wq_conv2 = (h16*)alloc((size_t)128 * 4032 * 2);
  h16* wq_att4  = (h16*)alloc((size_t)352 * 3168 * 2);
  h16* wq_conv3 = (h16*)alloc((size_t)128 * 352  * 2);

  int4*   dsi = (int4*)  alloc((size_t)B * 9 * HW1 * sizeof(int4));
  float4* dsw = (float4*)alloc((size_t)B * 9 * HW1 * sizeof(float4));
  int4*   dli = (int4*)  alloc((size_t)B * 9 * HW2 * sizeof(int4));
  float4* dlw = (float4*)alloc((size_t)B * 9 * HW2 * sizeof(float4));

  float* RA = (float*)alloc((size_t)B * 448 * HW2 * 4);  // buf2 / buf5
  float* RB = (float*)alloc((size_t)B * 448 * HW2 * 4);  // buf1,buf1b,hyb / buf3
  float* RC = (float*)alloc((size_t)B * 352 * HW2 * 4);  // buf4

  float* pooled  = (float*)alloc((size_t)B * 448 * 4);
  float* se_a    = (float*)alloc((size_t)B * 448 * 4);
  float* se_b    = (float*)alloc((size_t)B * 448 * 4);
  float* bn_mean = (float*)alloc((size_t)448 * 4);
  float* bn_inv  = (float*)alloc((size_t)448 * 4);

  // sub-aliases inside RB (dead before buf3 is written)
  float* buf1  = RB;                                   // [4,256,64,64]
  float* buf1b = RB + (size_t)B * 256 * HW1;           // [4,256,64,64]
  float* hyb   = buf1b + (size_t)B * 256 * HW1;        // [4,192,128,128]
  float* buf2  = RA;                                   // [4,448,128,128]
  float* buf3  = RB;                                   // [4,448,128,128]
  float* buf4  = RC;                                   // [4,352,128,128]
  float* buf5  = RA;                                   // [4,352,128,128] (buf2 dead)

  auto repack = [&](const float* w, h16* o, int M, int Cin, int khw) {
    int n = M * Cin * khw;
    repack_w_kernel<<<(n + 255) / 256, 256, 0, stream>>>(w, o, M, Cin, khw);
  };
  auto gemm = [&](const h16* A, const float* X, const float* bias, float* out,
                  const int4* di, const float4* dw, int M, int Cin, int khw,
                  int mode, int logw, int act, int Ctot, int c_off) {
    int HW = 1 << (2 * logw);
    dim3 g((unsigned)((B * HW) / 128), (unsigned)(M / 32));
    if (mode == MODE_CONV3 && logw == 6)
      wmma_conv_kernel<MODE_CONV3, 6><<<g, 256, 0, stream>>>(
          A, X, bias, out, di, dw, M, Cin, khw, act, Ctot, c_off);
    else if (mode == MODE_CONV3 && logw == 7)
      wmma_conv_kernel<MODE_CONV3, 7><<<g, 256, 0, stream>>>(
          A, X, bias, out, di, dw, M, Cin, khw, act, Ctot, c_off);
    else if (mode == MODE_DEF && logw == 6)
      wmma_conv_kernel<MODE_DEF, 6><<<g, 256, 0, stream>>>(
          A, X, bias, out, di, dw, M, Cin, khw, act, Ctot, c_off);
    else if (mode == MODE_DEF && logw == 7)
      wmma_conv_kernel<MODE_DEF, 7><<<g, 256, 0, stream>>>(
          A, X, bias, out, di, dw, M, Cin, khw, act, Ctot, c_off);
    else
      wmma_conv_kernel<MODE_CONV1, 7><<<g, 256, 0, stream>>>(
          A, X, bias, out, di, dw, M, Cin, khw, act, Ctot, c_off);
  };
  auto se = [&](float* x, int C, int HW, int Ctot, int c_off,
                const float* f1w, const float* f1b,
                const float* f2w, const float* f2b) {
    se_pool_kernel<<<B * C, 256, 0, stream>>>(x, pooled, C, HW, Ctot, c_off);
    int fcb = (B * C + 255) / 256;
    se_fc_kernel<<<fcb, 256, 0, stream>>>(pooled, f1w, f1b, se_a, B, C, ACT_RELU);
    se_fc_kernel<<<fcb, 256, 0, stream>>>(se_a,   f2w, f2b, se_b, B, C, ACT_SIG);
    long tot = (long)B * C * HW;
    se_apply_kernel<<<(unsigned)((tot + 255) / 256), 256, 0, stream>>>(
        x, se_b, C, HW, Ctot, c_off, tot);
  };

  // ---- weight repacks + deform descriptors ----
  repack(conv1_w, wq_conv1, 128, 256, 9);
  repack(aug1_w,  wq_aug1,  128, 256, 9);
  repack(a1_cw,   wq_att1,  256, 256, 9);
  repack(aug2_w,  wq_aug2,  64,  128, 9);
  repack(a2_cw,   wq_att2,  192, 192, 9);
  repack(a3_cw,   wq_att3,  448, 448, 9);
  repack(aug3_w,  wq_aug3,  224, 448, 9);
  repack(conv2_w, wq_conv2, 128, 448, 9);
  repack(a4_cw,   wq_att4,  352, 352, 9);
  repack(conv3_w, wq_conv3, 128, 352, 1);
  deform_desc_kernel<<<(B * 9 * HW1 + 255) / 256, 256, 0, stream>>>(
      offset_small, dsi, dsw, B, 64, 64);
  deform_desc_kernel<<<(B * 9 * HW2 + 255) / 256, 256, 0, stream>>>(
      offset_large, dli, dlw, B, 128, 128);

  // ---- stage 1 (64x64): conv1(elu) | aug1 -> concat -> att1(new_dem) ----
  gemm(wq_conv1, input_feature, conv1_b, buf1, dsi, dsw,
       128, 256, 9, MODE_CONV3, 6, ACT_ELU, 256, 0);
  gemm(wq_aug1, input_feature, aug1_b, buf1, dsi, dsw,
       128, 256, 9, MODE_DEF, 6, ACT_NONE, 256, 128);
  gemm(wq_att1, buf1, a1_cb, buf1b, dsi, dsw,
       256, 256, 9, MODE_CONV3, 6, ACT_RELU, 256, 0);
  se(buf1b, 256, HW1, 256, 0, a1_f1w, a1_f1b, a1_f2w, a1_f2b);
  {
    long tot = (long)B * 256 * HW2;
    upsample2x_kernel<<<(unsigned)((tot + 255) / 256), 256, 0, stream>>>(
        buf1b, buf2, B, 256, 64, 64, 448, 0);
  }

  // ---- stage 2 (128x128): aug2 | encoder -> hyb -> att2(new_dem) ----
  gemm(wq_aug2, encoder, aug2_b, hyb, dli, dlw,
       64, 128, 9, MODE_DEF, 7, ACT_NONE, 192, 0);
  {
    long tot = (long)B * 128 * HW2;
    copy_ch_kernel<<<(unsigned)((tot + 255) / 256), 256, 0, stream>>>(
        encoder, hyb, B, 128, HW2, 192, 64);
  }
  gemm(wq_att2, hyb, a2_cb, buf2, dli, dlw,
       192, 192, 9, MODE_CONV3, 7, ACT_RELU, 448, 256);
  se(buf2, 192, HW2, 448, 256, a2_f1w, a2_f1b, a2_f2w, a2_f2b);

  // ---- stage 3: att3 = conv -> BN(batch stats) -> relu -> SE ----
  gemm(wq_att3, buf2, a3_cb, buf3, dli, dlw,
       448, 448, 9, MODE_CONV3, 7, ACT_NONE, 448, 0);
  bn_stats_kernel<<<448, 256, 0, stream>>>(buf3, bn_mean, bn_inv, 448, HW2, B);
  {
    long tot = (long)B * 448 * HW2;
    bn_apply_relu_kernel<<<(unsigned)((tot + 255) / 256), 256, 0, stream>>>(
        buf3, bn_mean, bn_inv, a3_bng, a3_bnb, 448, HW2, tot);
  }
  se(buf3, 448, HW2, 448, 0, a3_f1w, a3_f1b, a3_f2w, a3_f2b);

  // ---- stage 4: conv2(elu) | aug3 -> concat -> att4(new_dem) ----
  gemm(wq_aug3, buf3, aug3_b, buf4, dli, dlw,
       224, 448, 9, MODE_DEF, 7, ACT_NONE, 352, 128);
  gemm(wq_conv2, buf3, conv2_b, buf4, dli, dlw,
       128, 448, 9, MODE_CONV3, 7, ACT_ELU, 352, 0);
  gemm(wq_att4, buf4, a4_cb, buf5, dli, dlw,
       352, 352, 9, MODE_CONV3, 7, ACT_RELU, 352, 0);
  se(buf5, 352, HW2, 352, 0, a4_f1w, a4_f1b, a4_f2w, a4_f2b);

  // ---- final 1x1 conv + elu -> d_out [4,128,128,128] ----
  gemm(wq_conv3, buf5, conv3_b, (float*)d_out, dli, dlw,
       128, 352, 1, MODE_CONV1, 7, ACT_ELU, 128, 0);
}
